// MultiHeadAttention_79869211836709
// MI455X (gfx1250) — matrix operand order repaired and checked by reading, not compiled, not run on hardware
//
#include <hip/hip_runtime.h>
#include <hip/hip_bf16.h>

// MHA for MI455X (gfx1250, wave32, WMMA). All GEMMs + attention run on
// v_wmma_f32_16x16x32_bf16. Tiles are staged into LDS with the CDNA5
// async-DMA path (global_load_async_to_lds_b128 / s_wait_asynccnt) and
// double-buffered so the next tile's DMA overlaps the current tile's WMMAs.
// Flash-style causal attention (no S x S score materialization).

typedef unsigned short u16;
typedef unsigned int   u32;
typedef unsigned long long u64;
typedef __attribute__((ext_vector_type(4)))  u32    u32x4;
typedef __attribute__((ext_vector_type(4)))  float  f32x4;
typedef __attribute__((ext_vector_type(16))) __bf16 v16bf;
typedef __attribute__((ext_vector_type(8)))  float  v8f;

#define B_  4
#define S_  2048
#define D_  1024
#define H_  16
#define HD_ 64

__device__ __forceinline__ u16 f2bf(float f) {
  u32 u = __float_as_uint(f);
  u += 0x7FFFu + ((u >> 16) & 1u);   // round-to-nearest-even
  return (u16)(u >> 16);
}

union FragU { u32x4 q[2]; v16bf v; };
union V8U   { u32x4 q; u16 s[8]; };

__device__ __forceinline__ v8f wmma_bf16(v16bf a, v16bf b, v8f c) {
  // (neg_a, A, neg_b, B, c_mod, C, reuse_a, reuse_b)
  return __builtin_amdgcn_wmma_f32_16x16x32_bf16(false, a, false, b,
                                                 (short)0, c, false, false);
}

// Async 16B DMA: memory[sbase + voff] -> LDS[lds]. GVS mode, ASYNCcnt-tracked.
__device__ __forceinline__ void async_b128(u32 lds, u32 voff, u64 sbase) {
  asm volatile("global_load_async_to_lds_b128 %0, %1, %2"
               :: "v"(lds), "v"(voff), "s"(sbase) : "memory");
}
__device__ __forceinline__ void wait_async0() {
  asm volatile("s_wait_asynccnt 0x0" ::: "memory");
}

// A fragment (16x32 bf16): lane L holds row = row_base + (L&15);
// lanes 0-15: K = {0..7, 16..23}; lanes 16-31: K = {8..15, 24..31} (ISA 7.12.2)
__device__ __forceinline__ v16bf load_frag_a(const u16* base, int ld,
                                             int row_base, int kbase, int lane) {
  const int hf = lane >> 4, mr = lane & 15;
  const u16* p = base + (row_base + mr) * ld + kbase + hf * 8;
  FragU u;
  u.q[0] = *(const u32x4*)p;
  u.q[1] = *(const u32x4*)(p + 16);
  return u.v;
}

// B fragment (32x16 bf16) from K-contiguous (N-major) storage:
// lane L holds col = col_base + (L&15); lanes 0-15: K = {0..7, 16..23}; lanes 16-31: K = {8..15, 24..31}
__device__ __forceinline__ v16bf load_frag_b(const u16* base, int ld,
                                             int col_base, int kbase, int lane) {
  const int hf = lane >> 4, nc = lane & 15;
  const u16* p = base + (col_base + nc) * ld + kbase + hf * 8;
  FragU u;
  u.q[0] = *(const u32x4*)p;
  u.q[1] = *(const u32x4*)(p + 16);
  return u.v;
}

// ---------------- precision conversion kernels ----------------

__global__ __launch_bounds__(256) void cvt_f32_bf16(const float* __restrict__ src,
                                                    u16* __restrict__ dst) {
  const size_t i = (size_t)blockIdx.x * 256 + threadIdx.x;
  f32x4 v = ((const f32x4*)src)[i];
  ((u32*)dst)[2 * i]     = (u32)f2bf(v.x) | ((u32)f2bf(v.y) << 16);
  ((u32*)dst)[2 * i + 1] = (u32)f2bf(v.z) | ((u32)f2bf(v.w) << 16);
}

// W [K=1024][N=1024] row-major f32 -> Wt [N][K] bf16 (B^T, K-contiguous)
__global__ __launch_bounds__(256) void cvt_transpose(const float* __restrict__ W,
                                                     u16* __restrict__ Wt) {
  const int idx = blockIdx.x * 256 + threadIdx.x;
  const int k = idx >> 10, n = idx & 1023;
  Wt[(size_t)n * D_ + k] = f2bf(W[idx]);
}

// ---------------- WMMA GEMM:  C[M,N] = A[M,K] * Bt[N,K]^T (+bias) ----------------
// 128x128 block tile, 8 waves (4M x 2N), each wave 32x64 = 2x4 WMMA frags.
// K-tile 32. Tiles DMA'd into double-buffered LDS with async-to-LDS; DMA of
// tile k+1 overlaps WMMA of tile k.

#define GT_ELEMS (128 * 40)          // one LDS tile (elements)
#define GT_BYTES (GT_ELEMS * 2)      // 10240 B

template <bool OUT_BF16, bool BIAS>
__global__ __launch_bounds__(256) void gemm_bf16_wmma(
    const u16* __restrict__ A, const u16* __restrict__ Bt, const int Kdim,
    u16* __restrict__ Cb, float* __restrict__ Cf,
    const float* __restrict__ bias, const int ldc) {
  __shared__ u16 As[2][GT_ELEMS];
  __shared__ u16 Bs[2][GT_ELEMS];

  const int tid = threadIdx.x;
  const int m0 = blockIdx.y * 128, n0 = blockIdx.x * 128;
  const int w = tid >> 5, lane = tid & 31, hf = lane >> 4, ml = lane & 15;
  const int wm = w & 3, wn = w >> 2;

  const u16* gA = A + (size_t)m0 * Kdim;
  const u16* gB = Bt + (size_t)n0 * Kdim;
  const u64 gAb = (u64)(size_t)gA;
  const u64 gBb = (u64)(size_t)gB;
  const u32 ldsA = (u32)(size_t)&As[0][0];
  const u32 ldsB = (u32)(size_t)&Bs[0][0];

  const int r0 = tid >> 2;          // row 0..63 (and +64)
  const int sc = (tid & 3) * 8;     // 8-elem (16B) segment in K

  const u32 lo0 = (u32)((r0 * 40 + sc) * 2);          // LDS byte offsets
  const u32 lo1 = lo0 + (u32)(64 * 40 * 2);

  // issue tile 0 DMA
  {
    const u32 v0 = (u32)((r0 * Kdim + sc) * 2);
    const u32 v1 = (u32)(((r0 + 64) * Kdim + sc) * 2);
    async_b128(ldsA + lo0, v0, gAb);
    async_b128(ldsA + lo1, v1, gAb);
    async_b128(ldsB + lo0, v0, gBb);
    async_b128(ldsB + lo1, v1, gBb);
  }

  v8f acc[2][4];
#pragma unroll
  for (int mi = 0; mi < 2; ++mi)
#pragma unroll
    for (int ni = 0; ni < 4; ++ni)
#pragma unroll
      for (int r = 0; r < 8; ++r) acc[mi][ni][r] = 0.f;

  int ib = 0;
  for (int k0 = 0; k0 < Kdim; k0 += 32, ib ^= 1) {
    wait_async0();        // this wave's DMA for tile k0 has landed in LDS
    __syncthreads();      // all waves' DMAs done; previous buffer consumed

    const int kn = k0 + 32;
    if (kn < Kdim) {      // kick off DMA for next tile into the other buffer
      const u32 bofs = (u32)((ib ^ 1) * GT_BYTES);
      const u32 v0 = (u32)((r0 * Kdim + kn + sc) * 2);
      const u32 v1 = (u32)(((r0 + 64) * Kdim + kn + sc) * 2);
      async_b128(ldsA + bofs + lo0, v0, gAb);
      async_b128(ldsA + bofs + lo1, v1, gAb);
      async_b128(ldsB + bofs + lo0, v0, gBb);
      async_b128(ldsB + bofs + lo1, v1, gBb);
    }

    const u16* curA = &As[ib][0];
    const u16* curB = &Bs[ib][0];
    v16bf af[2], bfv[4];
#pragma unroll
    for (int mi = 0; mi < 2; ++mi)
      af[mi] = load_frag_a(curA, 40, wm * 32 + mi * 16, 0, lane);
#pragma unroll
    for (int ni = 0; ni < 4; ++ni)
      bfv[ni] = load_frag_b(curB, 40, wn * 64 + ni * 16, 0, lane);

#pragma unroll
    for (int mi = 0; mi < 2; ++mi)
#pragma unroll
      for (int ni = 0; ni < 4; ++ni)
        acc[mi][ni] = wmma_bf16(af[mi], bfv[ni], acc[mi][ni]);
  }

  // C layout: VGPR r -> row (r + 8*hf), col = lane&15 within the 16x16 tile
#pragma unroll
  for (int mi = 0; mi < 2; ++mi) {
    const int row0 = m0 + wm * 32 + mi * 16 + hf * 8;
#pragma unroll
    for (int ni = 0; ni < 4; ++ni) {
      const int col = n0 + wn * 64 + ni * 16 + ml;
      const float bv = BIAS ? bias[col] : 0.f;
#pragma unroll
      for (int r = 0; r < 8; ++r) {
        const size_t off = (size_t)(row0 + r) * ldc + col;
        if (OUT_BF16) Cb[off] = f2bf(acc[mi][ni][r]);
        else          Cf[off] = acc[mi][ni][r] + bv;
      }
    }
  }
}

// ---------------- causal flash attention ----------------
// grid (S/64, B*H), block 128 (4 waves). Each wave owns 16 query rows.
// Key tiles of 32; K tile arrives via async-to-LDS DMA, V via register
// transpose. Online softmax in the WMMA C-layout; P bounced through
// per-wave LDS to re-enter the A-fragment layout for PV.

__global__ __launch_bounds__(128) void flash_attn(
    const u16* __restrict__ Q, const u16* __restrict__ K,
    const u16* __restrict__ V, u16* __restrict__ ctx) {
  __shared__ u16 Qs[64 * 72];      // [q][hd]
  __shared__ u16 Ks[32 * 72];      // [key][hd]   (B-layout for Q.K^T)
  __shared__ u16 Vt[64 * 40];      // [hd][key]   (B-layout for P.V)
  __shared__ u16 Ps[4 * 16 * 40];  // per-wave P tile [q][key]

  const int tid = threadIdx.x;
  const int w = tid >> 5, lane = tid & 31, hf = lane >> 4, ml = lane & 15;
  const int bh = blockIdx.y, bb = bh >> 4, hh = bh & 15;
  const int qbase = blockIdx.x * 64;
  const size_t hoff = (size_t)bb * S_ * D_ + (size_t)hh * HD_;
  const u16* Qg = Q + hoff;
  const u16* Kg = K + hoff;
  const u16* Vg = V + hoff;
  const u64 Kgb = (u64)(size_t)Kg;
  const u32 ldsK = (u32)(size_t)&Ks[0];

  for (int i = tid; i < 512; i += 128) {
    const int row = i >> 3, seg = (i & 7) * 8;
    u32x4 val = *(const u32x4*)(Qg + (size_t)(qbase + row) * D_ + seg);
    *(u32x4*)&Qs[row * 72 + seg] = val;
  }

  float mrow[8], lrow[8];
  v8f o[4];
#pragma unroll
  for (int r = 0; r < 8; ++r) { mrow[r] = -3.0e38f; lrow[r] = 0.f; }
#pragma unroll
  for (int ni = 0; ni < 4; ++ni)
#pragma unroll
    for (int r = 0; r < 8; ++r) o[ni][r] = 0.f;

  const int nkt = (qbase >> 5) + 2;   // causal: keys up to qbase+63
  for (int kt = 0; kt < nkt; ++kt) {
    const int kb = kt * 32;
    __syncthreads();                  // previous tile fully consumed
    // K tile (32x64) by async DMA: 256 x 16B, 2 per thread
    for (int i = tid; i < 256; i += 128) {
      const int row = i >> 3, seg = (i & 7) * 8;
      async_b128(ldsK + (u32)((row * 72 + seg) * 2),
                 (u32)(((kb + row) * D_ + seg) * 2), Kgb);
    }
    for (int i = tid; i < 256; i += 128) {   // V transposed into LDS
      const int row = i >> 3, seg = (i & 7) * 8;
      V8U val;
      val.q = *(const u32x4*)(Vg + (size_t)(kb + row) * D_ + seg);
#pragma unroll
      for (int j = 0; j < 8; ++j) Vt[(seg + j) * 40 + row] = val.s[j];
    }
    wait_async0();
    __syncthreads();

    // scores S = Q K^T : 16 q-rows x 32 keys, contract over HD=64 (2 WMMA steps)
    const v16bf a0 = load_frag_a(Qs, 72, w * 16, 0, lane);
    const v16bf a1 = load_frag_a(Qs, 72, w * 16, 32, lane);
    const v16bf bk00 = load_frag_b(Ks, 72, 0, 0, lane);
    const v16bf bk01 = load_frag_b(Ks, 72, 0, 32, lane);
    const v16bf bk10 = load_frag_b(Ks, 72, 16, 0, lane);
    const v16bf bk11 = load_frag_b(Ks, 72, 16, 32, lane);
    v8f s0, s1;
#pragma unroll
    for (int r = 0; r < 8; ++r) { s0[r] = 0.f; s1[r] = 0.f; }
    s0 = wmma_bf16(a0, bk00, s0);
    s1 = wmma_bf16(a0, bk10, s1);
    s0 = wmma_bf16(a1, bk01, s0);
    s1 = wmma_bf16(a1, bk11, s1);

    // online softmax; row stats reduced across the 16-lane half
    const int qrow0 = qbase + w * 16 + hf * 8;
    float pr0[8], pr1[8];
#pragma unroll
    for (int r = 0; r < 8; ++r) {
      const int qg = qrow0 + r;
      float x0 = (kb + ml <= qg)      ? s0[r] * 0.125f : -3.0e38f;
      float x1 = (kb + 16 + ml <= qg) ? s1[r] * 0.125f : -3.0e38f;
      float rm = fmaxf(x0, x1);
      rm = fmaxf(rm, __shfl_xor(rm, 1));
      rm = fmaxf(rm, __shfl_xor(rm, 2));
      rm = fmaxf(rm, __shfl_xor(rm, 4));
      rm = fmaxf(rm, __shfl_xor(rm, 8));
      const float mn = fmaxf(mrow[r], rm);
      const float corr = __expf(mrow[r] - mn);
      const float p0 = __expf(x0 - mn);
      const float p1 = __expf(x1 - mn);
      float ps = p0 + p1;
      ps += __shfl_xor(ps, 1);
      ps += __shfl_xor(ps, 2);
      ps += __shfl_xor(ps, 4);
      ps += __shfl_xor(ps, 8);
      lrow[r] = lrow[r] * corr + ps;
      mrow[r] = mn;
      pr0[r] = p0; pr1[r] = p1;
#pragma unroll
      for (int ni = 0; ni < 4; ++ni) o[ni][r] *= corr;
    }

    // P (C-layout) -> LDS -> A-fragment layout; wave-local, LDS is in-order
    u16* pw = Ps + w * 16 * 40;
#pragma unroll
    for (int r = 0; r < 8; ++r) {
      pw[(hf * 8 + r) * 40 + ml]      = f2bf(pr0[r]);
      pw[(hf * 8 + r) * 40 + 16 + ml] = f2bf(pr1[r]);
    }
    const v16bf pa = load_frag_a(pw, 40, 0, 0, lane);
#pragma unroll
    for (int ni = 0; ni < 4; ++ni) {
      const v16bf vb = load_frag_b(Vt, 40, ni * 16, 0, lane);
      o[ni] = wmma_bf16(pa, vb, o[ni]);
    }
  }

  u16* cg = ctx + hoff;
#pragma unroll
  for (int r = 0; r < 8; ++r) {
    const int qg = qbase + w * 16 + hf * 8 + r;
    const float inv = 1.0f / lrow[r];
#pragma unroll
    for (int ni = 0; ni < 4; ++ni)
      cg[(size_t)qg * D_ + ni * 16 + ml] = f2bf(o[ni][r] * inv);
  }
}

// ---------------- launch ----------------

extern "C" void kernel_launch(void* const* d_in, const int* in_sizes, int n_in,
                              void* d_out, int out_size, void* d_ws, size_t ws_size,
                              hipStream_t stream) {
  const float* x  = (const float*)d_in[0];
  const float* Wq = (const float*)d_in[1];
  const float* Wk = (const float*)d_in[2];
  const float* Wv = (const float*)d_in[3];
  const float* Wo = (const float*)d_in[4];
  const float* bo = (const float*)d_in[5];
  float* out = (float*)d_out;

  const size_t NX = (size_t)B_ * S_ * D_;  // 8,388,608
  const size_t NW = (size_t)D_ * D_;       // 1,048,576

  u16* xb  = (u16*)d_ws;
  u16* WqT = xb  + NX;
  u16* WkT = WqT + NW;
  u16* WvT = WkT + NW;
  u16* WoT = WvT + NW;
  u16* Qb  = WoT + NW;
  u16* Kb  = Qb  + NX;
  u16* Vb  = Kb  + NX;
  u16* Cx  = Vb  + NX;   // total ~93 MB of workspace

  cvt_f32_bf16<<<(unsigned)(NX / 4 / 256), 256, 0, stream>>>(x, xb);
  cvt_transpose<<<(unsigned)(NW / 256), 256, 0, stream>>>(Wq, WqT);
  cvt_transpose<<<(unsigned)(NW / 256), 256, 0, stream>>>(Wk, WkT);
  cvt_transpose<<<(unsigned)(NW / 256), 256, 0, stream>>>(Wv, WvT);
  cvt_transpose<<<(unsigned)(NW / 256), 256, 0, stream>>>(Wo, WoT);

  dim3 gg(D_ / 128, (B_ * S_) / 128);  // (8, 64)
  gemm_bf16_wmma<true, false><<<gg, 256, 0, stream>>>(xb, WqT, D_, Qb, nullptr, nullptr, D_);
  gemm_bf16_wmma<true, false><<<gg, 256, 0, stream>>>(xb, WkT, D_, Kb, nullptr, nullptr, D_);
  gemm_bf16_wmma<true, false><<<gg, 256, 0, stream>>>(xb, WvT, D_, Vb, nullptr, nullptr, D_);

  flash_attn<<<dim3(S_ / 64, B_ * H_), 128, 0, stream>>>(Qb, Kb, Vb, Cx);

  gemm_bf16_wmma<false, true><<<gg, 256, 0, stream>>>(Cx, WoT, D_, nullptr, out, bo, D_);
}
